// TransformerBlock_55645596287146
// MI455X (gfx1250) — matrix operand order repaired and checked by reading, not compiled, not run on hardware
//
#include <hip/hip_runtime.h>
#include <math.h>

// ---------------- problem constants ----------------
#define B_    2
#define T_    2048
#define C_    1024
#define H_    16
#define KV_   4
#define DH_   64
#define NREP_ (H_ / KV_)
#define FF_   4096
#define E_    8
#define WIN_  1024
#define NTOK  (B_ * T_)
#define LB_W  0.01f
#define Z_W   0.001f

typedef __attribute__((ext_vector_type(16))) __bf16 v16bf;
typedef __attribute__((ext_vector_type(8)))  float  v8f;

union U8 { uint4 u; __bf16 h[8]; };

// contiguous 16 bf16 -> fragment (used for B fragments: K = half*16 + i)
__device__ __forceinline__ v16bf frag_contig16(const __bf16* p) {
    U8 a, b;
    a.u = *(const uint4*)p;
    b.u = *(const uint4*)(p + 8);
    v16bf f;
#pragma unroll
    for (int i = 0; i < 8; ++i) { f[i] = a.h[i]; f[i + 8] = b.h[i]; }
    return f;
}

// A fragment (16-bit 16x32 layout): contiguous run [kk+16*half .. +15]
__device__ __forceinline__ v16bf frag_a(const __bf16* rowp, int kk, int half) {
    U8 a, b;
    a.u = *(const uint4*)(rowp + kk + 16 * half);
    b.u = *(const uint4*)(rowp + kk + 16 * half + 8);
    v16bf f;
#pragma unroll
    for (int i = 0; i < 8; ++i) { f[i] = a.h[i]; f[i + 8] = b.h[i]; }
    return f;
}

__device__ __forceinline__ v16bf frag_zero() {
    v16bf f;
#pragma unroll
    for (int i = 0; i < 16; ++i) f[i] = (__bf16)0.0f;
    return f;
}

// ---------------- GEMM: C[M,N] = A_bf16[M,K] @ B_f32[K,N], WMMA bf16 ----------------
#define GM_F32   0
#define GM_BF16  1
#define GM_RESID 2
#define GM_ACC   3

// Block: 256 threads = 8 waves (2 along M x 4 along N), block tile 32M x 64N,
// K stepped by 64 (two WMMAs per barrier pair). B tile staged f32->bf16 into
// LDS transposed [n][k], stride 72 keeps fragment reads 16B-aligned for b128.
// LDB and MODE are compile-time so staging offsets fold into load immediates.
template <int LDB, int MODE>
__global__ __launch_bounds__(256)
void gemm_bf16_wmma(const __bf16* __restrict__ A, int lda,
                    const float*  __restrict__ Bw,
                    float* Cf, __bf16* Cb, int ldc, int K,
                    const float* __restrict__ resid,
                    const float* __restrict__ alpha,
                    const float* __restrict__ rowscale, int rs_stride) {
    __shared__ __bf16 lds_b[64 * 72];

    const int tid   = threadIdx.x;
    const int lane  = tid & 31;
    const int wid   = tid >> 5;
    const int half  = lane >> 4;
    const int l16   = lane & 15;
    const int m0    = blockIdx.x * 32;
    const int n0    = blockIdx.y * 64;
    const int wm    = (wid >> 2) * 16;   // 2 waves along M
    const int wn    = (wid & 3) * 16;    // 4 waves along N

    // staging: thread owns 16 consecutive k of one column n
    const int n_  = tid & 63;
    const int k0  = (tid >> 6) * 16;

    const __bf16* aptr = A + (size_t)(m0 + wm + l16) * lda;
    const float*  bsrc = Bw + (size_t)k0 * LDB + n0 + n_;
    __bf16*       lwr  = &lds_b[n_ * 72 + k0];            // 16B aligned
    const __bf16* brd  = &lds_b[(wn + l16) * 72 + half * 16];

    v8f acc = {};

    for (int kk = 0; kk < K; kk += 64) {
        // 1) clause of 16 global loads (immediate offsets j*LDB*4)
        float tmp[16];
#pragma unroll
        for (int j = 0; j < 16; ++j)
            tmp[j] = bsrc[(size_t)j * LDB];
        // prefetch next K-tile (4 per thread; the block covers every 4th of the 64 rows)
        if (kk + 64 < K) {
#pragma unroll
            for (int j = 0; j < 16; j += 4)
                __builtin_prefetch(bsrc + (size_t)(64 + j) * LDB, 0, 1);
        }
        // 2) convert + two b128 LDS stores (16 contiguous bf16 in [n][k] tile)
        union { uint4 q[2]; __bf16 h[16]; } pk;
#pragma unroll
        for (int j = 0; j < 16; ++j)
            pk.h[j] = (__bf16)tmp[j];
        *(uint4*)lwr       = pk.q[0];
        *((uint4*)lwr + 1) = pk.q[1];
        __syncthreads();

        // 3) fragments + two WMMAs
        v16bf af0 = frag_a(aptr, 0, half);
        v16bf af1 = frag_a(aptr, 32, half);
        v16bf bf0 = frag_contig16(brd);
        v16bf bf1 = frag_contig16(brd + 32);
        acc = __builtin_amdgcn_wmma_f32_16x16x32_bf16(false, af0, false, bf0,
                                                      (short)0, acc, false, false);
        acc = __builtin_amdgcn_wmma_f32_16x16x32_bf16(false, af1, false, bf1,
                                                      (short)0, acc, false, false);
        __syncthreads();

        aptr += 64;
        bsrc += (size_t)64 * LDB;
    }

    const int nIdx  = n0 + wn + l16;
    const int mBase = m0 + wm + 8 * half;
#pragma unroll
    for (int r = 0; r < 8; ++r) {
        size_t idx = (size_t)(mBase + r) * ldc + nIdx;
        float v = acc[r];
        if (MODE == GM_F32) {
            Cf[idx] = v;
        } else if (MODE == GM_BF16) {
            Cb[idx] = (__bf16)v;
        } else if (MODE == GM_RESID) {
            Cf[idx] = resid[idx] + alpha[0] * v;
        } else { // GM_ACC
            float sc = rowscale ? rowscale[(size_t)(mBase + r) * rs_stride] : 1.0f;
            Cf[idx] += sc * v;
        }
    }
}

// ---------------- RMSNorm: f32 in -> bf16 out ----------------
__global__ __launch_bounds__(256)
void rmsnorm_kernel(const float* __restrict__ x, const float* __restrict__ w,
                    __bf16* __restrict__ out) {
    __shared__ float red[8];
    const int row = blockIdx.x;
    const float* xr = x + (size_t)row * C_;
    float ss = 0.f;
    for (int c = threadIdx.x; c < C_; c += 256) { float v = xr[c]; ss += v * v; }
#pragma unroll
    for (int off = 16; off > 0; off >>= 1) ss += __shfl_xor(ss, off, 32);
    if ((threadIdx.x & 31) == 0) red[threadIdx.x >> 5] = ss;
    __syncthreads();
    float tot = 0.f;
#pragma unroll
    for (int i = 0; i < 8; ++i) tot += red[i];
    const float rms = rsqrtf(tot / (float)C_ + 1e-6f);
    __bf16* orow = out + (size_t)row * C_;
    for (int c = threadIdx.x; c < C_; c += 256)
        orow[c] = (__bf16)(xr[c] * rms * w[c]);
}

// ---------------- RoPE: f32 in -> bf16 out ----------------
__global__ void rope_kernel(const float* __restrict__ in, __bf16* __restrict__ out,
                            int heads, long total) {
    long idx = (long)blockIdx.x * blockDim.x + threadIdx.x;
    if (idx >= total) return;
    int d = (int)(idx % DH_);
    long row = idx / (heads * DH_);          // b*T + t
    int t = (int)(row % T_);
    float f   = (float)(d & 31) / 32.0f;
    float ang = (float)t * __powf(10000.0f, -f);
    float s, c;
    __sincosf(ang, &s, &c);
    float u   = in[idx];
    float rot = (d < 32) ? -in[idx + 32] : in[idx - 32];
    out[idx] = (__bf16)(u * c + rot * s);
}

// ---------------- flash-style sliding-window GQA attention (1 wave / q-tile) ------
__global__ __launch_bounds__(32)
void attn_kernel(const __bf16* __restrict__ Q, const __bf16* __restrict__ Kb,
                 const __bf16* __restrict__ Vb, __bf16* __restrict__ O) {
    __shared__ __bf16 lds_p[16 * 16];
    __shared__ __bf16 lds_vt[64 * 16];   // [d][k]

    const int bh  = blockIdx.x;
    const int b   = bh / H_;
    const int h   = bh % H_;
    const int kvh = h / NREP_;
    const int q0  = blockIdx.y * 16;

    const int lane = threadIdx.x & 31;
    const int half = lane >> 4;
    const int l16  = lane & 15;

    // Q A-fragments (two K-steps over DH=64)
    const __bf16* qrow = Q + (size_t)(b * T_ + q0 + l16) * (H_ * DH_) + h * DH_;
    v16bf qa0 = frag_a(qrow, 0, half);
    v16bf qa1 = frag_a(qrow, 32, half);

    float mrow[8], lrow[8];
#pragma unroll
    for (int r = 0; r < 8; ++r) { mrow[r] = -1e30f; lrow[r] = 0.f; }
    v8f acc[4] = {{}, {}, {}, {}};

    int kt_lo = (q0 - WIN_) > 0 ? (q0 - WIN_) >> 4 : 0;
    int kt_hi = q0 >> 4;

    for (int kt = kt_lo; kt <= kt_hi; ++kt) {
        // K^T B-fragments: lane column = key l16, contiguous d-run
        const __bf16* krow = Kb + (size_t)(b * T_ + kt * 16 + l16) * (KV_ * DH_) + kvh * DH_;
        v16bf kb0 = frag_contig16(krow + 0  + half * 16);
        v16bf kb1 = frag_contig16(krow + 32 + half * 16);

        v8f s_acc = {};
        s_acc = __builtin_amdgcn_wmma_f32_16x16x32_bf16(false, qa0, false, kb0,
                                                        (short)0, s_acc, false, false);
        s_acc = __builtin_amdgcn_wmma_f32_16x16x32_bf16(false, qa1, false, kb1,
                                                        (short)0, s_acc, false, false);

        // stage V tile transposed: lane = key row l16, d-range half*32..+31
        {
            const __bf16* vrow = Vb + (size_t)(b * T_ + kt * 16 + l16) * (KV_ * DH_)
                                 + kvh * DH_ + half * 32;
            union { uint4 u[4]; __bf16 h[32]; } vt;
            const uint4* vp = (const uint4*)vrow;
#pragma unroll
            for (int j = 0; j < 4; ++j) vt.u[j] = vp[j];
#pragma unroll
            for (int j = 0; j < 32; ++j)
                lds_vt[(half * 32 + j) * 16 + l16] = vt.h[j];
        }

        // online softmax update (row reductions over 16-lane half-wave)
        const int ki = kt * 16 + l16;
#pragma unroll
        for (int r = 0; r < 8; ++r) {
            int qi = q0 + r + 8 * half;
            float sv = s_acc[r] * 0.125f;
            bool msk = (ki > qi) || (qi > ki + WIN_);
            sv = msk ? -1e30f : sv;
            float rm = sv;
#pragma unroll
            for (int off = 1; off < 16; off <<= 1) rm = fmaxf(rm, __shfl_xor(rm, off, 16));
            float mnew = fmaxf(mrow[r], rm);
            float p  = msk ? 0.f : __expf(sv - mnew);
            float rs = p;
#pragma unroll
            for (int off = 1; off < 16; off <<= 1) rs += __shfl_xor(rs, off, 16);
            float cf = __expf(mrow[r] - mnew);
            lrow[r] = lrow[r] * cf + rs;
            mrow[r] = mnew;
#pragma unroll
            for (int dt = 0; dt < 4; ++dt) acc[dt][r] *= cf;
            lds_p[(r + 8 * half) * 16 + l16] = (__bf16)p;
        }

        // P as A-fragment: half 0 holds K = i (cols 0..15), half 1 (K 16..31) zero
        v16bf pa;
        {
            U8 a; a.u = *(const uint4*)&lds_p[l16 * 16];
            U8 c; c.u = *(const uint4*)&lds_p[l16 * 16 + 8];
#pragma unroll
            for (int i = 0; i < 8; ++i) { pa[i] = half ? (__bf16)0.0f : a.h[i]; pa[i + 8] = half ? (__bf16)0.0f : c.h[i]; }
        }

        // P @ V : 4 output d-tiles; B half1 (K 16..31) is zero
#pragma unroll
        for (int dt = 0; dt < 4; ++dt) {
            v16bf vb;
            if (half == 0) vb = frag_contig16(&lds_vt[(dt * 16 + l16) * 16]);
            else           vb = frag_zero();
            acc[dt] = __builtin_amdgcn_wmma_f32_16x16x32_bf16(false, pa, false, vb,
                                                              (short)0, acc[dt], false, false);
        }
    }

    // normalize + store context bf16 [NTOK, H*DH]
#pragma unroll
    for (int dt = 0; dt < 4; ++dt) {
#pragma unroll
        for (int r = 0; r < 8; ++r) {
            int m = q0 + r + 8 * half;
            int d = dt * 16 + l16;
            float v = acc[dt][r] / lrow[r];
            O[(size_t)(b * T_ + m) * (H_ * DH_) + h * DH_ + d] = (__bf16)v;
        }
    }
}

// ---------------- silu(h1)*h2 -> bf16 ----------------
__global__ void silu_mul_kernel(const float* __restrict__ h1, const float* __restrict__ h2,
                                __bf16* __restrict__ out, long n) {
    long i = (long)blockIdx.x * blockDim.x + threadIdx.x;
    if (i >= n) return;
    float a = h1[i];
    float s = a / (1.0f + __expf(-a));
    out[i] = (__bf16)(s * h2[i]);
}

// ---------------- gate logits: xf_bf16 [NTOK,C] @ gate_w [C,E] ----------------
__global__ __launch_bounds__(64)
void gate_kernel(const __bf16* __restrict__ xf, const float* __restrict__ gw,
                 float* __restrict__ logits) {
    __shared__ float gred[2][E_];
    const int t = blockIdx.x;
    const int tid = threadIdx.x;
    float acc[E_];
#pragma unroll
    for (int e = 0; e < E_; ++e) acc[e] = 0.f;
    const __bf16* xr = xf + (size_t)t * C_;
    for (int c = tid; c < C_; c += 64) {
        float xv = (float)xr[c];
#pragma unroll
        for (int e = 0; e < E_; ++e) acc[e] += xv * gw[c * E_ + e];
    }
#pragma unroll
    for (int e = 0; e < E_; ++e) {
        float v = acc[e];
#pragma unroll
        for (int off = 16; off > 0; off >>= 1) v += __shfl_xor(v, off, 32);
        acc[e] = v;
    }
    if ((tid & 31) == 0) {
#pragma unroll
        for (int e = 0; e < E_; ++e) gred[tid >> 5][e] = acc[e];
    }
    __syncthreads();
    if (tid == 0) {
#pragma unroll
        for (int e = 0; e < E_; ++e) logits[(size_t)t * E_ + e] = gred[0][e] + gred[1][e];
    }
}

// ---------------- router: top-2 combine weights + aux-loss stats ----------------
__global__ void router_kernel(const float* __restrict__ logits, float* __restrict__ cw,
                              float* __restrict__ stats /* [0..7]=load, [8]=zsum */) {
    int t = blockIdx.x * blockDim.x + threadIdx.x;
    if (t >= NTOK) return;
    float l[E_];
#pragma unroll
    for (int e = 0; e < E_; ++e) l[e] = logits[(size_t)t * E_ + e];
    float mx = l[0];
#pragma unroll
    for (int e = 1; e < E_; ++e) mx = fmaxf(mx, l[e]);
    float sum = 0.f;
    float p[E_];
#pragma unroll
    for (int e = 0; e < E_; ++e) { p[e] = __expf(l[e] - mx); sum += p[e]; }
    float inv = 1.0f / sum;
#pragma unroll
    for (int e = 0; e < E_; ++e) atomicAdd(&stats[e], p[e] * inv);
    float lse = mx + logf(sum);
    atomicAdd(&stats[E_], lse * lse);

    int i1 = 0;
#pragma unroll
    for (int e = 1; e < E_; ++e) if (l[e] > l[i1]) i1 = e;
    int i2 = (i1 == 0) ? 1 : 0;
#pragma unroll
    for (int e = 0; e < E_; ++e) if (e != i1 && l[e] > l[i2]) i2 = e;
    float w1 = 1.0f / (1.0f + __expf(l[i2] - l[i1]));
    cw[(size_t)t * E_ + i1] = w1;
    cw[(size_t)t * E_ + i2] = 1.0f - w1;
}

// ---------------- finalize ----------------
__global__ void final_out_kernel(const float* __restrict__ h, const float* __restrict__ moe,
                                 const float* __restrict__ alpha, float* __restrict__ out, long n) {
    long i = (long)blockIdx.x * blockDim.x + threadIdx.x;
    if (i >= n) return;
    out[i] = h[i] + alpha[0] * moe[i];
}

__global__ void aux_final_kernel(const float* __restrict__ stats, float* __restrict__ out_aux) {
    float sl = 0.f;
    for (int e = 0; e < E_; ++e) sl += stats[e];
    float lb = 0.f;
    for (int e = 0; e < E_; ++e) { float f = stats[e] / sl; lb += f * f; }
    lb *= (float)E_;
    float z = stats[E_] / (float)NTOK;
    out_aux[0] = LB_W * lb + Z_W * z;
}

__global__ void zero_kernel(float* p, long n) {
    long i = (long)blockIdx.x * blockDim.x + threadIdx.x;
    if (i < n) p[i] = 0.f;
}

// ---------------- launcher ----------------
static inline dim3 g1(long n, int b) { return dim3((unsigned)((n + b - 1) / b)); }

extern "C" void kernel_launch(void* const* d_in, const int* in_sizes, int n_in,
                              void* d_out, int out_size, void* d_ws, size_t ws_size,
                              hipStream_t stream) {
    const float* x      = (const float*)d_in[0];
    const float* ln1w   = (const float*)d_in[1];
    const float* ln2w   = (const float*)d_in[2];
    const float* wq     = (const float*)d_in[3];
    const float* wk     = (const float*)d_in[4];
    const float* wv     = (const float*)d_in[5];
    const float* wo     = (const float*)d_in[6];
    const float* sw1    = (const float*)d_in[7];
    const float* sw2    = (const float*)d_in[8];
    const float* sw3    = (const float*)d_in[9];
    const float* ew1    = (const float*)d_in[10];
    const float* ew2    = (const float*)d_in[11];
    const float* ew3    = (const float*)d_in[12];
    const float* gatew  = (const float*)d_in[13];
    const float* a_attn = (const float*)d_in[14];
    const float* a_moe  = (const float*)d_in[15];
    float* out = (float*)d_out;

    // workspace layout (256B aligned)
    char* base = (char*)d_ws;
    size_t off = 0;
    auto alloc = [&](size_t bytes) -> void* {
        void* p = base + off;
        off = (off + bytes + 255) & ~(size_t)255;
        return p;
    };
    __bf16* xn1_bf = (__bf16*)alloc((size_t)NTOK * C_ * 2);
    float*  qf     = (float*) alloc((size_t)NTOK * (H_ * DH_) * 4);
    float*  kf     = (float*) alloc((size_t)NTOK * (KV_ * DH_) * 4);
    __bf16* q_bf   = (__bf16*)alloc((size_t)NTOK * (H_ * DH_) * 2);
    __bf16* k_bf   = (__bf16*)alloc((size_t)NTOK * (KV_ * DH_) * 2);
    __bf16* v_bf   = (__bf16*)alloc((size_t)NTOK * (KV_ * DH_) * 2);
    __bf16* ctx_bf = (__bf16*)alloc((size_t)NTOK * C_ * 2);
    float*  hbuf   = (float*) alloc((size_t)NTOK * C_ * 4);
    __bf16* xf_bf  = (__bf16*)alloc((size_t)NTOK * C_ * 2);
    float*  hid1   = (float*) alloc((size_t)NTOK * FF_ * 4);
    float*  hid2   = (float*) alloc((size_t)NTOK * FF_ * 4);
    __bf16* hid_bf = (__bf16*)alloc((size_t)NTOK * FF_ * 2);
    float*  moe    = (float*) alloc((size_t)NTOK * C_ * 4);
    float*  logits = (float*) alloc((size_t)NTOK * E_ * 4);
    float*  cw     = (float*) alloc((size_t)NTOK * E_ * 4);
    float*  stats  = (float*) alloc(16 * 4);

    const long nC  = (long)NTOK * C_;
    const long nFF = (long)NTOK * FF_;

    // zero accumulators
    zero_kernel<<<g1(nC, 256), 256, 0, stream>>>(moe, nC);
    zero_kernel<<<g1((long)NTOK * E_, 256), 256, 0, stream>>>(cw, (long)NTOK * E_);
    zero_kernel<<<1, 16, 0, stream>>>(stats, 9);

    // 1) rmsnorm1
    rmsnorm_kernel<<<NTOK, 256, 0, stream>>>(x, ln1w, xn1_bf);

    // 2) QKV projections (WMMA GEMMs)
    gemm_bf16_wmma<1024, GM_F32><<<dim3(NTOK / 32, (H_ * DH_) / 64), 256, 0, stream>>>(
        xn1_bf, C_, wq, qf, nullptr, H_ * DH_, C_, nullptr, nullptr, nullptr, 0);
    gemm_bf16_wmma<256, GM_F32><<<dim3(NTOK / 32, (KV_ * DH_) / 64), 256, 0, stream>>>(
        xn1_bf, C_, wk, kf, nullptr, KV_ * DH_, C_, nullptr, nullptr, nullptr, 0);
    gemm_bf16_wmma<256, GM_BF16><<<dim3(NTOK / 32, (KV_ * DH_) / 64), 256, 0, stream>>>(
        xn1_bf, C_, wv, nullptr, v_bf, KV_ * DH_, C_, nullptr, nullptr, nullptr, 0);

    // 3) RoPE
    rope_kernel<<<g1((long)NTOK * H_ * DH_, 256), 256, 0, stream>>>(qf, q_bf, H_, (long)NTOK * H_ * DH_);
    rope_kernel<<<g1((long)NTOK * KV_ * DH_, 256), 256, 0, stream>>>(kf, k_bf, KV_, (long)NTOK * KV_ * DH_);

    // 4) sliding-window GQA attention
    attn_kernel<<<dim3(B_ * H_, T_ / 16), 32, 0, stream>>>(q_bf, k_bf, v_bf, ctx_bf);

    // 5) output projection + ReZero residual: h = x + alpha_attn * (ctx @ wo)
    gemm_bf16_wmma<1024, GM_RESID><<<dim3(NTOK / 32, C_ / 64), 256, 0, stream>>>(
        ctx_bf, H_ * DH_, wo, hbuf, nullptr, C_, H_ * DH_, x, a_attn, nullptr, 0);

    // 6) rmsnorm2
    rmsnorm_kernel<<<NTOK, 256, 0, stream>>>(hbuf, ln2w, xf_bf);

    // 7) router
    gate_kernel<<<NTOK, 64, 0, stream>>>(xf_bf, gatew, logits);
    router_kernel<<<g1(NTOK, 256), 256, 0, stream>>>(logits, cw, stats);

    // 8) shared expert SwiGLU -> moe accumulator
    gemm_bf16_wmma<4096, GM_F32><<<dim3(NTOK / 32, FF_ / 64), 256, 0, stream>>>(
        xf_bf, C_, sw1, hid1, nullptr, FF_, C_, nullptr, nullptr, nullptr, 0);
    gemm_bf16_wmma<4096, GM_F32><<<dim3(NTOK / 32, FF_ / 64), 256, 0, stream>>>(
        xf_bf, C_, sw2, hid2, nullptr, FF_, C_, nullptr, nullptr, nullptr, 0);
    silu_mul_kernel<<<g1(nFF, 256), 256, 0, stream>>>(hid1, hid2, hid_bf, nFF);
    gemm_bf16_wmma<1024, GM_ACC><<<dim3(NTOK / 32, C_ / 64), 256, 0, stream>>>(
        hid_bf, FF_, sw3, moe, nullptr, C_, FF_, nullptr, nullptr, nullptr, 0);

    // 9) routed experts (dense scan, combine weight folded into w3 epilogue)
    for (int e = 0; e < E_; ++e) {
        const float* w1e = ew1 + (size_t)e * C_ * FF_;
        const float* w2e = ew2 + (size_t)e * C_ * FF_;
        const float* w3e = ew3 + (size_t)e * FF_ * C_;
        gemm_bf16_wmma<4096, GM_F32><<<dim3(NTOK / 32, FF_ / 64), 256, 0, stream>>>(
            xf_bf, C_, w1e, hid1, nullptr, FF_, C_, nullptr, nullptr, nullptr, 0);
        gemm_bf16_wmma<4096, GM_F32><<<dim3(NTOK / 32, FF_ / 64), 256, 0, stream>>>(
            xf_bf, C_, w2e, hid2, nullptr, FF_, C_, nullptr, nullptr, nullptr, 0);
        silu_mul_kernel<<<g1(nFF, 256), 256, 0, stream>>>(hid1, hid2, hid_bf, nFF);
        gemm_bf16_wmma<1024, GM_ACC><<<dim3(NTOK / 32, C_ / 64), 256, 0, stream>>>(
            hid_bf, FF_, w3e, moe, nullptr, C_, FF_, nullptr, nullptr, cw + e, E_);
    }

    // 10) out = h + alpha_moe * moe ; aux scalar
    final_out_kernel<<<g1(nC, 256), 256, 0, stream>>>(hbuf, moe, a_moe, out, nC);
    aux_final_kernel<<<1, 1, 0, stream>>>(stats, out + nC);
}
